// SelfAttention_6940667150757
// MI455X (gfx1250) — compile-verified, matrix operand order checked by reading, not run on hardware
//
#include <hip/hip_runtime.h>

// Problem constants (match reference)
#define B_  4
#define S_  2048
#define DM  2048
#define NH  16
#define HD  128
#define QTS (S_/16)   // 128 query tiles per (b,h)

typedef __bf16 bf16x16 __attribute__((ext_vector_type(16)));
typedef float  f32x8   __attribute__((ext_vector_type(8)));
typedef int    v4i     __attribute__((vector_size(16)));   // matches builtin param type

union BFVec {
  bf16x16 v;
  unsigned int u[8];
  unsigned short s[16];
};

// f32 -> bf16 round-to-nearest-even (bit math; no __bf16 arithmetic needed)
__device__ __forceinline__ unsigned short f2bf(float f) {
  unsigned int u = __float_as_uint(f);
  return (unsigned short)((u + 0x7FFFu + ((u >> 16) & 1u)) >> 16);
}

// 16-bit A/B operand K-offset of dword pair v (0..7) within a 32-wide K chunk.
// CDNA5 ISA: VGPR v holds K = (v>>2)*16 + half*8 + 2*(v&3) + {0,1}
__device__ __forceinline__ int koff(int v, int half) {
  return ((v >> 2) << 4) | (half << 3) | ((v & 3) << 1);
}

__device__ __forceinline__ f32x8 wmma_bf16(bf16x16 a, bf16x16 b, f32x8 c) {
  return __builtin_amdgcn_wmma_f32_16x16x32_bf16(
      /*neg_a=*/false, a, /*neg_b=*/false, b,
      /*c_mod=*/(short)0, c, /*reuse_a=*/false, /*reuse_b=*/false);
}

// ---- 16B copy into LDS: async DMA path if the gfx1250 builtins exist, else
//      global_load_b128 + ds_store_b128 fallback (both compile-safe). ----
#if defined(__has_builtin)
#if __has_builtin(__builtin_amdgcn_global_load_async_to_lds_b128) && \
    __has_builtin(__builtin_amdgcn_s_wait_asynccnt)
#define USE_ASYNC_LDS 1
#endif
#endif
#ifndef USE_ASYNC_LDS
#define USE_ASYNC_LDS 0
#endif

__device__ __forceinline__ void cp16(unsigned short* lds_dst, const unsigned short* gsrc) {
#if USE_ASYNC_LDS
  __builtin_amdgcn_global_load_async_to_lds_b128(
      (__attribute__((address_space(1))) v4i*)(gsrc),
      (__attribute__((address_space(3))) v4i*)(lds_dst), 0, 0);
#else
  *(uint4*)lds_dst = *(const uint4*)gsrc;
#endif
}

__device__ __forceinline__ void wait_cp() {
#if USE_ASYNC_LDS
  __builtin_amdgcn_s_wait_asynccnt(0);
#endif
}

// ---------------------------------------------------------------------------
// Kernel 1: RoPE(K) + RMSNorm(Q,K) (+ fold 1/sqrt(dh) into Q) + bf16 convert.
// One wave32 per (b, s, h); lane handles 4 consecutive head-dim elements.
// V stored transposed per head: Vt[((b*NH+h)*HD + d)*S + s]
// ---------------------------------------------------------------------------
__global__ __launch_bounds__(256) void qkv_prep(
    const float* __restrict__ q, const float* __restrict__ k,
    const float* __restrict__ v,
    unsigned short* __restrict__ Qb, unsigned short* __restrict__ Kb,
    unsigned short* __restrict__ Vt) {
  int wid  = blockIdx.x * 8 + (threadIdx.x >> 5);   // global wave id
  int lane = threadIdx.x & 31;
  int h = wid & (NH - 1);
  int s = (wid >> 4) & (S_ - 1);
  int b = wid >> 15;
  size_t base = ((size_t)(b * S_ + s)) * DM + h * HD + lane * 4;

  // ---- Q: RMS norm, fold attention scale ----
  float qv[4];
  float ss = 0.f;
  #pragma unroll
  for (int i = 0; i < 4; ++i) { qv[i] = q[base + i]; ss += qv[i] * qv[i]; }
  #pragma unroll
  for (int m = 1; m < 32; m <<= 1) ss += __shfl_xor(ss, m);
  float r = rsqrtf(ss * (1.0f / HD) + 1e-6f) * 0.08838834764831845f; // * 1/sqrt(128)
  #pragma unroll
  for (int i = 0; i < 4; ++i) Qb[base + i] = f2bf(qv[i] * r);

  // ---- K: RoPE then RMS norm ----
  float kv[4];
  #pragma unroll
  for (int i = 0; i < 4; ++i) kv[i] = k[base + i];
  #pragma unroll
  for (int p = 0; p < 2; ++p) {
    int pi = lane * 2 + p;                       // pair index 0..63
    // inv_freq = 10000^-(pi/64) = exp(-pi * ln(10000)/64)
    float inv_freq = __expf(-(float)pi * (9.210340371976184f / 64.0f));
    float ang = (float)s * inv_freq;
    float c, sn;
    __sincosf(ang, &sn, &c);
    float x1 = kv[2 * p], x2 = kv[2 * p + 1];
    kv[2 * p]     = x1 * c - x2 * sn;
    kv[2 * p + 1] = x1 * sn + x2 * c;
  }
  ss = 0.f;
  #pragma unroll
  for (int i = 0; i < 4; ++i) ss += kv[i] * kv[i];
  #pragma unroll
  for (int m = 1; m < 32; m <<= 1) ss += __shfl_xor(ss, m);
  r = rsqrtf(ss * (1.0f / HD) + 1e-6f);
  #pragma unroll
  for (int i = 0; i < 4; ++i) Kb[base + i] = f2bf(kv[i] * r);

  // ---- V: transpose per head for contiguous B-operand gathers ----
  #pragma unroll
  for (int i = 0; i < 4; ++i) {
    int d = lane * 4 + i;
    Vt[((size_t)((b * NH + h) * HD + d)) * S_ + s] = f2bf(v[base + i]);
  }
}

// ---------------------------------------------------------------------------
// Kernel 2: w_proj -> bf16 (row-major [n][k], matching B-operand row gathers)
// ---------------------------------------------------------------------------
__global__ __launch_bounds__(256) void w_prep(const float* __restrict__ w,
                                              unsigned short* __restrict__ Wb, int n) {
  int i = blockIdx.x * 256 + threadIdx.x;
  if (i < n) Wb[i] = f2bf(w[i]);
}

// ---------------------------------------------------------------------------
// Kernel 3: causal flash attention, LDS-tiled.
// One 256-thread block (8 waves) owns 8 consecutive 16-query tiles of one
// (b,h) and shares 32-key K/V tiles via double-buffered LDS (async copies
// when available). Each wave computes S^T = K * Q^T so that P^T's C-layout
// is already the A-operand layout for P*V (no transposes). Softmax in f32.
// ---------------------------------------------------------------------------
__global__ __launch_bounds__(256) void flash_attn(
    const unsigned short* __restrict__ Qb, const unsigned short* __restrict__ Kb,
    const unsigned short* __restrict__ Vt, unsigned short* __restrict__ Yb) {
  __shared__ __align__(16) unsigned short Ksh[2][32 * HD];   // [buf][key][d]   8KB each
  __shared__ __align__(16) unsigned short Vsh[2][HD * 32];   // [buf][d][key]   8KB each

  int tid  = threadIdx.x;
  int warp = tid >> 5;
  int lane = tid & 31;
  int bid = blockIdx.x;             // grid = B*NH*(QTS/8) = 1024
  int qb = bid & 15;                // 128-query block within sequence
  int h  = (bid >> 4) & (NH - 1);
  int b  = bid >> 8;

  int qt = qb * 8 + warp;           // this wave's 16-query tile
  int mbase  = qt * 16;
  int row  = lane & 15;
  int half = lane >> 4;
  int q_glob = mbase + row;

  const f32x8 zero = (f32x8)0.0f;

  // Q rows in per-lane row format (lane holds query row `row`)
  BFVec qa[4];
  const unsigned int* qrow =
      (const unsigned int*)(Qb + ((size_t)(b * S_ + mbase + row)) * DM + h * HD);
  #pragma unroll
  for (int c = 0; c < 4; ++c)
    #pragma unroll
    for (int v = 0; v < 8; ++v)
      qa[c].u[v] = qrow[(32 * c + koff(v, half)) >> 1];

  f32x8 o[8];
  #pragma unroll
  for (int t = 0; t < 8; ++t) o[t] = zero;
  float m_run = -3.0e38f, l_run = 0.0f;

  const unsigned short* Kg0 = Kb + ((size_t)(b * S_)) * DM + h * HD;        // + key*DM
  const unsigned short* Vg0 = Vt + ((size_t)((b * NH + h) * HD)) * S_;      // + d*S + key

  int nkb = qb * 4 + 4;             // block covers keys [0, qb*128+128)

  // issue copies for key block 0 into buffer 0
  {
    #pragma unroll
    for (int i = 0; i < 2; ++i) {
      int c = tid * 2 + i;          // 0..511 sixteen-byte chunks
      int krow = c >> 4, kcol = (c & 15) << 3;
      cp16(&Ksh[0][c * 8], Kg0 + (size_t)krow * DM + kcol);
      int d = c >> 2, seg = (c & 3) << 3;
      cp16(&Vsh[0][c * 8], Vg0 + (size_t)d * S_ + seg);
    }
  }

  for (int kb = 0; kb < nkb; ++kb) {
    int cur = kb & 1;
    int k0 = kb * 32;

    wait_cp();          // my async copies for buf[cur] are done
    __syncthreads();    // everyone's copies are done; prior reads of buf[cur^1] done

    // prefetch next key block into the other buffer (overlaps with compute)
    if (kb + 1 < nkb) {
      int kn = (kb + 1) * 32;
      #pragma unroll
      for (int i = 0; i < 2; ++i) {
        int c = tid * 2 + i;
        int krow = c >> 4, kcol = (c & 15) << 3;
        cp16(&Ksh[cur ^ 1][c * 8], Kg0 + (size_t)(kn + krow) * DM + kcol);
        int d = c >> 2, seg = (c & 3) << 3;
        cp16(&Vsh[cur ^ 1][c * 8], Vg0 + (size_t)d * S_ + kn + seg);
      }
    }

    // waves fully past their causal frontier just keep the barriers going
    if (k0 <= mbase + 15) {
      const unsigned short* Kc = Ksh[cur];
      const unsigned short* Vc = Vsh[cur];
      const unsigned int* kr0 = (const unsigned int*)(Kc + (size_t)row * HD);
      const unsigned int* kr1 = (const unsigned int*)(Kc + (size_t)(16 + row) * HD);

      // S^T tiles: D[key][query]; lane = query, VGPR r = key r + 8*half
      f32x8 st0 = zero, st1 = zero;
      #pragma unroll
      for (int c = 0; c < 4; ++c) {
        BFVec ka0, ka1;
        #pragma unroll
        for (int v = 0; v < 8; ++v) {
          int idx = (32 * c + koff(v, half)) >> 1;
          ka0.u[v] = kr0[idx];
          ka1.u[v] = kr1[idx];
        }
        st0 = wmma_bf16(ka0.v, qa[c].v, st0);
        st1 = wmma_bf16(ka1.v, qa[c].v, st1);
      }

      // causal mask + per-query block max (8 in-lane + cross-half shfl)
      float s0[8], s1[8];
      float mblk = -3.0e38f;
      #pragma unroll
      for (int r2 = 0; r2 < 8; ++r2) {
        int key0 = k0 + r2 + 8 * half;
        s0[r2] = (key0      <= q_glob) ? st0[r2] : -3.0e38f;
        s1[r2] = (key0 + 16 <= q_glob) ? st1[r2] : -3.0e38f;
        mblk = fmaxf(mblk, fmaxf(s0[r2], s1[r2]));
      }
      mblk = fmaxf(mblk, __shfl_xor(mblk, 16));
      float mnew = fmaxf(m_run, mblk);
      float corr = __expf(m_run - mnew);

      float p0[8], p1[8];
      float psum = 0.f;
      #pragma unroll
      for (int r2 = 0; r2 < 8; ++r2) {
        p0[r2] = __expf(s0[r2] - mnew);
        p1[r2] = __expf(s1[r2] - mnew);
        psum += p0[r2] + p1[r2];
      }
      psum += __shfl_xor(psum, 16);
      l_run = l_run * corr + psum;
      m_run = mnew;

      // rescale O: broadcast per-query corr into per-row (VGPR) form
      f32x8 fvec;
      #pragma unroll
      for (int r2 = 0; r2 < 8; ++r2) fvec[r2] = __shfl(corr, r2 + 8 * half);
      #pragma unroll
      for (int t = 0; t < 8; ++t) o[t] *= fvec;

      // Pack P^T (already in A-operand order): elems 0..7 = keys half*8+j,
      // elems 8..15 = keys 16+half*8+(j-8) -> exactly p0[], p1[]
      BFVec pa;
      #pragma unroll
      for (int r2 = 0; r2 < 8; ++r2) { pa.s[r2] = f2bf(p0[r2]); pa.s[8 + r2] = f2bf(p1[r2]); }

      // O += P * V ; B operand from transposed V tile (contiguous key pairs)
      #pragma unroll
      for (int t = 0; t < 8; ++t) {
        const unsigned int* vc = (const unsigned int*)(Vc + (size_t)(t * 16 + row) * 32);
        BFVec vv;
        #pragma unroll
        for (int v = 0; v < 8; ++v) vv.u[v] = vc[koff(v, half) >> 1];
        o[t] = wmma_bf16(pa.v, vv.v, o[t]);
      }
    }

    __syncthreads();   // all reads of buf[cur] done before it is re-filled
  }

  // normalize by 1/l (broadcast per query into row form) and store bf16 Y
  float inv = 1.0f / l_run;
  f32x8 linv;
  #pragma unroll
  for (int r2 = 0; r2 < 8; ++r2) linv[r2] = __shfl(inv, r2 + 8 * half);
  #pragma unroll
  for (int t = 0; t < 8; ++t)
    #pragma unroll
    for (int r2 = 0; r2 < 8; ++r2) {
      int m = mbase + r2 + 8 * half;
      Yb[((size_t)(b * S_ + m)) * DM + h * HD + t * 16 + row] = f2bf(o[t][r2] * linv[r2]);
    }
}

// ---------------------------------------------------------------------------
// Kernel 4: out = Y @ W^T. One wave computes a 16x64 f32 tile (4 WMMA accums,
// A fragment reused 4x per K chunk). K loop: 2048/32 = 64 iterations.
// ---------------------------------------------------------------------------
__global__ __launch_bounds__(256) void proj_gemm(
    const unsigned short* __restrict__ Yb, const unsigned short* __restrict__ Wb,
    float* __restrict__ out) {
  int wid  = blockIdx.x * 8 + (threadIdx.x >> 5);
  int lane = threadIdx.x & 31;
  int nt4 = wid & 31;          // 64-wide N block (2048/64 = 32)
  int mt  = wid >> 5;          // 16-row M tile (8192/16 = 512)
  int row  = lane & 15;
  int half = lane >> 4;
  int mbase = mt * 16, nbase = nt4 * 64;

  const f32x8 zero = (f32x8)0.0f;
  f32x8 acc[4];
  #pragma unroll
  for (int t = 0; t < 4; ++t) acc[t] = zero;

  const unsigned int* arow = (const unsigned int*)(Yb + (size_t)(mbase + row) * DM);
  const unsigned int* brow[4];
  #pragma unroll
  for (int t = 0; t < 4; ++t)
    brow[t] = (const unsigned int*)(Wb + (size_t)(nbase + t * 16 + row) * DM);

  for (int kc = 0; kc < DM / 32; ++kc) {
    BFVec av;
    #pragma unroll
    for (int v = 0; v < 8; ++v) av.u[v] = arow[(kc * 32 + koff(v, half)) >> 1];
    #pragma unroll
    for (int t = 0; t < 4; ++t) {
      BFVec bv;
      #pragma unroll
      for (int v = 0; v < 8; ++v) bv.u[v] = brow[t][(kc * 32 + koff(v, half)) >> 1];
      acc[t] = wmma_bf16(av.v, bv.v, acc[t]);
    }
  }

  #pragma unroll
  for (int t = 0; t < 4; ++t)
    #pragma unroll
    for (int r2 = 0; r2 < 8; ++r2)
      out[(size_t)(mbase + r2 + 8 * half) * DM + nbase + t * 16 + row] = acc[t][r2];
}

// ---------------------------------------------------------------------------
extern "C" void kernel_launch(void* const* d_in, const int* in_sizes, int n_in,
                              void* d_out, int out_size, void* d_ws, size_t ws_size,
                              hipStream_t stream) {
  const float* q = (const float*)d_in[0];
  const float* k = (const float*)d_in[1];
  const float* v = (const float*)d_in[2];
  const float* w = (const float*)d_in[3];
  float* out = (float*)d_out;

  unsigned short* wsp = (unsigned short*)d_ws;
  const size_t NTOK = (size_t)B_ * S_ * DM;   // 16.7M elems per tensor
  unsigned short* Qb = wsp;                   // bf16 normed+scaled Q
  unsigned short* Kb = Qb + NTOK;             // bf16 rope+normed K
  unsigned short* Vt = Kb + NTOK;             // bf16 V transposed per head
  unsigned short* Yb = Vt + NTOK;             // bf16 attention output
  unsigned short* Wb = Yb + NTOK;             // bf16 w_proj
  (void)ws_size; (void)in_sizes; (void)n_in; (void)out_size;

  // 1) preprocess: 131072 waves, 8 per block
  qkv_prep<<<(B_ * S_ * NH) / 8, 256, 0, stream>>>(q, k, v, Qb, Kb, Vt);
  // 2) weight convert
  w_prep<<<(DM * DM + 255) / 256, 256, 0, stream>>>(w, Wb, DM * DM);
  // 3) flash attention: 1024 blocks x 8 waves (8 q-tiles of one (b,h) each)
  flash_attn<<<B_ * NH * (QTS / 8), 256, 0, stream>>>(Qb, Kb, Vt, Yb);
  // 4) projection GEMM: 512*32 = 16384 waves, 8 per block
  proj_gemm<<<((B_ * S_ / 16) * (DM / 64)) / 8, 256, 0, stream>>>(Yb, Wb, out);
}
